// ResidualBlock_13365938225630
// MI455X (gfx1250) — compile-verified, matrix operand order checked by reading, not run on hardware
//
#include <hip/hip_runtime.h>

typedef __bf16 bf16;
typedef __attribute__((ext_vector_type(4)))  bf16  v4bf;
typedef __attribute__((ext_vector_type(8)))  bf16  v8bf;
typedef __attribute__((ext_vector_type(16))) bf16  v16bf;
typedef __attribute__((ext_vector_type(4)))  float v4f;
typedef __attribute__((ext_vector_type(8)))  float v8f;

#define C_CH 256
#define KOFF 27
#define BN_EPS 1e-4f
#define LDS_ROW 528     // 512B row + 16B pad -> 4-bank rotation per row, conflict-free b128 reads
#define MROWS 96        // points per block (6 WMMA M-tiles)
#define GCHUNKS ((MROWS * 32) / 256)   // 12 gather chunks per thread

#if defined(__has_builtin)
#  if __has_builtin(__builtin_amdgcn_sched_barrier)
#    define SCHED_FENCE() __builtin_amdgcn_sched_barrier(0)
#  endif
#endif
#ifndef SCHED_FENCE
#  define SCHED_FENCE()
#endif

// ---------------- BN helpers ----------------

__global__ void zero_stats_kernel(float* __restrict__ stats) {
    stats[threadIdx.x] = 0.0f;
    stats[threadIdx.x + C_CH] = 0.0f;
}

// zero the sentinel row hbuf[n][*] used for missing neighbors
__global__ void zero_row_kernel(bf16* __restrict__ row) {
    row[threadIdx.x] = (bf16)0.0f;
}

__global__ __launch_bounds__(256) void bn_stats_kernel(const float* __restrict__ x,
                                                       float* __restrict__ stats, int n) {
    const int c = threadIdx.x;           // one channel per thread, coalesced row reads
    float s = 0.0f, s2 = 0.0f;
    for (int row = blockIdx.x; row < n; row += gridDim.x) {
        float v = x[(size_t)row * C_CH + c];
        s += v;
        s2 += v * v;
    }
    atomicAdd(&stats[c], s);
    atomicAdd(&stats[C_CH + c], s2);
}

__global__ void bn_finalize_kernel(const float* __restrict__ stats,
                                   const float* __restrict__ gamma,
                                   const float* __restrict__ beta,
                                   float* __restrict__ ss, int n) {
    const int c = threadIdx.x;
    const float inv_n = 1.0f / (float)n;
    const float mean = stats[c] * inv_n;
    const float var  = stats[C_CH + c] * inv_n - mean * mean;
    const float sc   = rsqrtf(var + BN_EPS) * gamma[c];
    ss[c]        = sc;
    ss[C_CH + c] = beta[c] - mean * sc;
}

// y = bf16(relu(x*scale + shift)), 4 elements/thread
__global__ __launch_bounds__(256) void bn_apply_kernel(const float* __restrict__ x,
                                                       const float* __restrict__ ss,
                                                       bf16* __restrict__ y, int total) {
    const int i = (blockIdx.x * 256 + threadIdx.x) * 4;
    if (i >= total) return;
    v4f v  = *(const v4f*)(x + i);
    const int c0 = i & (C_CH - 1);
    v4f sc = *(const v4f*)(ss + c0);
    v4f sh = *(const v4f*)(ss + C_CH + c0);
    v4bf o;
#pragma unroll
    for (int j = 0; j < 4; ++j) {
        float t = v[j] * sc[j] + sh[j];
        t = t > 0.0f ? t : 0.0f;
        o[j] = (bf16)t;
    }
    *(v4bf*)(y + i) = o;
}

// Wt[k][cout][cin] = bf16(W[k][cin][cout])
__global__ __launch_bounds__(256) void wt_kernel(const float* __restrict__ W,
                                                 bf16* __restrict__ Wt) {
    const int gid = blockIdx.x * 256 + threadIdx.x;
    if (gid >= KOFF * C_CH * C_CH) return;
    const int k  = gid >> 16;
    const int r  = gid & 65535;
    const int co = r >> 8;
    const int ci = r & 255;
    Wt[gid] = (bf16)W[(k << 16) + (ci << 8) + co];
}

// ---------------- submanifold conv via WMMA ----------------
// Block: 256 threads = 8 waves. Block tile: 96 points x 256 couts.
// Wave w owns couts [w*32, w*32+32) across ALL 96 points:
//   6 M-tiles x 2 N-tiles = 12 v8f accumulators. Each B fragment is loaded
//   once per block and reused by 6 WMMAs.
__global__ __launch_bounds__(256) void subconv_kernel(const bf16* __restrict__ h,
                                                      const bf16* __restrict__ Wt,
                                                      const int* __restrict__ nbr,
                                                      const float* __restrict__ residual,
                                                      float* __restrict__ out, int n) {
    __shared__ __align__(16) unsigned char ldsA[MROWS * LDS_ROW];   // 50,688 B

    const int tid      = threadIdx.x;
    const int wave     = tid >> 5;
    const int lane     = tid & 31;
    const int laneLow  = lane & 15;
    const int hi       = lane >> 4;          // half-wave select
    const int coutBase = wave * 32;          // 32 couts per wave
    const int rowBase  = blockIdx.x * MROWS;

    v8f acc[12] = {};

    for (int k = 0; k < KOFF; ++k) {
        // --- gather 96 neighbor rows into LDS; missing -> zero row h[n] ---
        // Phase 1: issue all 12 global loads (clause, all in flight).
        uint4 g[GCHUNKS];
#pragma unroll
        for (int it = 0; it < GCHUNKS; ++it) {
            const int chunk = tid + it * 256;               // 96 rows x 32 chunks of 16B
            const int r  = chunk >> 5;
            const int cp = chunk & 31;
            const int row = rowBase + r;
            int idx = n;                                    // sentinel zero row
            if (row < n) {
                const int t = nbr[row * KOFF + k];
                idx = (t >= 0) ? t : n;
            }
            g[it] = *(const uint4*)((const unsigned char*)h + (size_t)idx * (C_CH * 2) + cp * 16);
        }
        // Phase 2: stream into LDS (loads complete in order -> staged waits).
#pragma unroll
        for (int it = 0; it < GCHUNKS; ++it) {
            const int chunk = tid + it * 256;
            const int r  = chunk >> 5;
            const int cp = chunk & 31;
            *(uint4*)(&ldsA[r * LDS_ROW + cp * 16]) = g[it];
        }
        __syncthreads();

        const bf16* wk = Wt + (size_t)k * (C_CH * C_CH);
#pragma unroll
        for (int kc = 0; kc < 8; ++kc) {
            const int kinB = kc * 32;
            // B fragments for this wave's 2 cout tiles (loaded once, reused x6)
            v16bf B[2];
#pragma unroll
            for (int t = 0; t < 2; ++t) {
                const bf16* bptr = wk + (size_t)(coutBase + t * 16 + laneLow) * C_CH
                                      + kinB + hi * 8;
                v8bf blo = *(const v8bf*)bptr;
                v8bf bhi = *(const v8bf*)(bptr + 16);
                B[t] = __builtin_shufflevector(blo, bhi,
                    0, 1, 2, 3, 4, 5, 6, 7, 8, 9, 10, 11, 12, 13, 14, 15);
            }
            // Preload ALL 6 A fragments into distinct registers.
            v16bf A[6];
#pragma unroll
            for (int p = 0; p < 6; ++p) {
                const unsigned char* aptr =
                    &ldsA[(p * 16 + laneLow) * LDS_ROW + (kinB + hi * 8) * 2];
                v8bf alo = *(const v8bf*)aptr;
                v8bf ahi = *(const v8bf*)(aptr + 32);
                A[p] = __builtin_shufflevector(alo, ahi,
                    0, 1, 2, 3, 4, 5, 6, 7, 8, 9, 10, 11, 12, 13, 14, 15);
            }
            // Scheduling fence: loads above may NOT sink past this point, so
            // all 12 ds_loads clause together with staged waits, instead of
            // ds_load x2 -> s_wait_dscnt 0 -> wmma x2 serialization.
            SCHED_FENCE();
#pragma unroll
            for (int p = 0; p < 6; ++p) {
                acc[p * 2 + 0] = __builtin_amdgcn_wmma_f32_16x16x32_bf16(
                    false, A[p], false, B[0], (short)0, acc[p * 2 + 0], false, false);
                acc[p * 2 + 1] = __builtin_amdgcn_wmma_f32_16x16x32_bf16(
                    false, A[p], false, B[1], (short)0, acc[p * 2 + 1], false, false);
            }
        }
        __syncthreads();
    }

    // C/D layout: VGPR r, lanes 0-15 -> M=r, lanes 16-31 -> M=8+r; N = laneLow
#pragma unroll
    for (int p = 0; p < 6; ++p) {
#pragma unroll
        for (int t = 0; t < 2; ++t) {
#pragma unroll
            for (int r = 0; r < 8; ++r) {
                const int row = rowBase + p * 16 + r + 8 * hi;
                if (row < n) {
                    const int col = coutBase + t * 16 + laneLow;
                    const size_t o = (size_t)row * C_CH + col;
                    float v = acc[p * 2 + t][r];
                    if (residual) v += residual[o];
                    out[o] = v;
                }
            }
        }
    }
}

// ---------------- host launcher ----------------
extern "C" void kernel_launch(void* const* d_in, const int* in_sizes, int n_in,
                              void* d_out, int out_size, void* d_ws, size_t ws_size,
                              hipStream_t stream) {
    const float* feat   = (const float*)d_in[0];
    // d_in[1] = pos (unused)
    const int*   nbr    = (const int*)d_in[2];
    const float* W1     = (const float*)d_in[3];
    const float* gamma1 = (const float*)d_in[4];
    const float* beta1  = (const float*)d_in[5];
    const float* W2     = (const float*)d_in[6];
    const float* gamma2 = (const float*)d_in[7];
    const float* beta2  = (const float*)d_in[8];

    const int n = in_sizes[0] / C_CH;
    const size_t nC = (size_t)n * C_CH;
    const size_t wElems = (size_t)KOFF * C_CH * C_CH;
    const size_t hbytes = (nC + C_CH) * 2;                 // n+1 rows of bf16 (sentinel)

    char* ws = (char*)d_ws;
    bf16*  hbuf  = (bf16*)ws;                              // (n+1)*C bf16 (h1 then h2)
    float* t1    = (float*)(ws + hbytes);                  // conv1 output, f32
    bf16*  wt1   = (bf16*)(ws + hbytes + nC * 4);          // transposed bf16 weights
    bf16*  wt2   = wt1 + wElems;
    float* stats = (float*)((char*)(wt2 + wElems));        // 2*C f32
    float* ss    = stats + 2 * C_CH;                       // scale/shift, 2*C f32

    const int convGrid  = (n + MROWS - 1) / MROWS;
    const int wGrid     = (int)((wElems + 255) / 256);
    const int applyGrid = (int)((nC / 4 + 255) / 256);

    // weight transpose + bf16 convert, sentinel zero row (independent of BN chain)
    wt_kernel<<<wGrid, 256, 0, stream>>>(W1, wt1);
    wt_kernel<<<wGrid, 256, 0, stream>>>(W2, wt2);
    zero_row_kernel<<<1, 256, 0, stream>>>(hbuf + nC);

    // BN1 + ReLU + bf16 -> hbuf
    zero_stats_kernel<<<1, 256, 0, stream>>>(stats);
    bn_stats_kernel<<<512, 256, 0, stream>>>(feat, stats, n);
    bn_finalize_kernel<<<1, 256, 0, stream>>>(stats, gamma1, beta1, ss, n);
    bn_apply_kernel<<<applyGrid, 256, 0, stream>>>(feat, ss, hbuf, (int)nC);

    // conv1 -> t1 (f32)
    subconv_kernel<<<convGrid, 256, 0, stream>>>(hbuf, wt1, nbr, nullptr, t1, n);

    // BN2 + ReLU + bf16 -> hbuf
    zero_stats_kernel<<<1, 256, 0, stream>>>(stats);
    bn_stats_kernel<<<512, 256, 0, stream>>>(t1, stats, n);
    bn_finalize_kernel<<<1, 256, 0, stream>>>(stats, gamma2, beta2, ss, n);
    bn_apply_kernel<<<applyGrid, 256, 0, stream>>>(t1, ss, hbuf, (int)nC);

    // conv2 + residual -> out
    subconv_kernel<<<convGrid, 256, 0, stream>>>(hbuf, wt2, nbr, feat, (float*)d_out, n);
}
